// BahdanauAttention_76252849373643
// MI455X (gfx1250) — matrix operand order repaired and checked by reading, not compiled, not run on hardware
//
#include <hip/hip_runtime.h>

typedef __attribute__((ext_vector_type(16))) _Float16 v16h;
typedef __attribute__((ext_vector_type(2)))  __fp16   v2hp;   // cvt_pkrtz result type
typedef __attribute__((ext_vector_type(8)))  float    v8f;

#define NEG_INF   (-1e9f)
#define N_ROWS    65536     // 64*32*32
#define CDIM      512
#define ADIM      512
#define ROWS_BLK  64
#define LDS_STRIDE 36       // 32 f32 payload + 4 dword pad (bank-conflict-free frag reads)

// ---- CDNA5 feature detection (guarded so both toolchains compile) ----
#if __has_builtin(__builtin_amdgcn_global_load_async_to_lds_b128) && \
    __has_builtin(__builtin_amdgcn_s_wait_asynccnt)
#define ATTN_ASYNC_COPY 1
typedef int v4i __attribute__((ext_vector_type(4)));
typedef __attribute__((address_space(1))) v4i* gptr_v4i;   // global (device) int4*
typedef __attribute__((address_space(3))) v4i* lptr_v4i;   // LDS int4*
#else
#define ATTN_ASYNC_COPY 0
#endif

#if __has_builtin(__builtin_amdgcn_tanhf)
#define HW_TANH(x) __builtin_amdgcn_tanhf(x)
#else
#define HW_TANH(x) tanhf(x)
#endif

// ---------------- zero-fill ----------------
__global__ void k_zero(float* __restrict__ p, int n) {
  int i = blockIdx.x * blockDim.x + threadIdx.x;
  if (i < n) p[i] = 0.f;
}

// -------- pack W1 (f32 [512,512] row-major, K x N) into f16 WMMA B-fragment layout -----
// fragment index f = kt*32 + nt   (kt = k/32, nt = n/16)
// within fragment: lane = (n%16) + (((k%32)&8) ? 16 : 0); half = ((k%32)&7) + (((k%32)>=16) ? 8 : 0)
__global__ __launch_bounds__(256) void k_pack_w1(const float* __restrict__ W1,
                                                 _Float16* __restrict__ W1p) {
  int i = blockIdx.x * 256 + threadIdx.x;
  if (i >= CDIM * ADIM) return;
  int k = i >> 9, n = i & 511;
  int kt = k >> 5, kk = k & 31;
  int nt = n >> 4, nn = n & 15;
  int lane = nn + ((kk & 8) << 1);
  int half = (kk & 7) + ((kk & 16) >> 1);
  W1p[((((kt << 5) + nt) * 32 + lane) << 4) + half] = (_Float16)W1[i];
}

// ---------------- h_proj = hidden @ W2 + b2  -> [64, 512] ----------------
__global__ __launch_bounds__(256) void k_hproj(const float* __restrict__ hidden,
                                               const float* __restrict__ W2,
                                               const float* __restrict__ b2,
                                               float* __restrict__ hproj) {
  int b = blockIdx.x;
  int a0 = threadIdx.x * 2;
  float acc0 = 0.f, acc1 = 0.f;
  for (int k = 0; k < 512; ++k) {
    float hk = hidden[b * 512 + k];
    float2 w = *(const float2*)(W2 + (size_t)k * 512 + a0);
    acc0 += hk * w.x;
    acc1 += hk * w.y;
  }
  hproj[b * 512 + a0]     = acc0 + b2[a0];
  hproj[b * 512 + a0 + 1] = acc1 + b2[a0 + 1];
}

// stage one 64x32 f32 tile (row stride CDIM) into LDS (row stride LDS_STRIDE dwords)
// 512 chunks of 16B; thread handles chunks {tid, 256+tid} -> coalesced 16B/lane
__device__ __forceinline__ void stage_tile(const float* __restrict__ g,
                                           float* __restrict__ ldsbase, int tid) {
#pragma unroll
  for (int i = 0; i < 2; ++i) {
    int c = i * 256 + tid;
    int row = c >> 3, part = c & 7;
    const float* gp = g + (size_t)row * CDIM + part * 4;
    float* lp = ldsbase + row * LDS_STRIDE + part * 4;
#if ATTN_ASYNC_COPY
    float* gpm = const_cast<float*>(gp);
    __builtin_amdgcn_global_load_async_to_lds_b128((gptr_v4i)gpm, (lptr_v4i)lp,
                                                   /*offset=*/0, /*cpol=*/0);
#else
    float4 v = *(const float4*)gp;
    *(float4*)lp = v;
#endif
  }
}

__device__ __forceinline__ void stage_wait() {
#if ATTN_ASYNC_COPY
  __builtin_amdgcn_s_wait_asynccnt(0);
#endif
}

// ---------------- fused: score = tanh(F@W1 + b1 + hproj) . V  (WMMA f16 -> f32) --------
__global__ __launch_bounds__(256) void k_score(const float* __restrict__ feat,
                                               const _Float16* __restrict__ W1p,
                                               const float* __restrict__ b1,
                                               const float* __restrict__ hproj,
                                               const float* __restrict__ V,
                                               const float* __restrict__ bV,
                                               float* __restrict__ score) {
  __shared__ float ldsA[2][ROWS_BLK][LDS_STRIDE];  // double-buffered f32 tile
  __shared__ float ldsSum[ROWS_BLK];

  const int tid    = threadIdx.x;
  const int wave   = tid >> 5;
  const int lane   = tid & 31;
  const int laneHi = (lane >> 4) & 1;
  const int laneN  = lane & 15;
  const int row0   = blockIdx.x * ROWS_BLK;
  const int batch  = row0 >> 10;            // 1024 rows per batch, 64 | 1024

  if (tid < ROWS_BLK) ldsSum[tid] = 0.f;

  v8f acc[4][4];                            // [row-tile][n-tile]
#pragma unroll
  for (int rt = 0; rt < 4; ++rt)
#pragma unroll
    for (int i = 0; i < 4; ++i)
#pragma unroll
      for (int e = 0; e < 8; ++e) acc[rt][i][e] = 0.f;

  const float* gbase = feat + (size_t)row0 * CDIM;

  // prologue: stage K-tile 0 into buffer 0
  stage_tile(gbase, &ldsA[0][0][0], tid);
  stage_wait();
  __syncthreads();

  for (int kt = 0; kt < CDIM / 32; ++kt) {
    const int cur = kt & 1;

    // issue copy of next K-tile into the other buffer (overlaps with WMMA below)
    if (kt < CDIM / 32 - 1)
      stage_tile(gbase + (kt + 1) * 32, &ldsA[1 - cur][0][0], tid);

    // B fragments for this K-tile (packed layout -> coalesced 32B/lane)
    v16h bf[4];
#pragma unroll
    for (int i = 0; i < 4; ++i) {
      int nt = (wave << 2) + i;
      bf[i] = *(const v16h*)(W1p + ((((size_t)kt * 32 + nt) * 32 + lane) << 4));
    }

    const int fo = laneHi << 3;             // f32 offset selecting this half's K-pairs
#pragma unroll
    for (int rt = 0; rt < 4; ++rt) {
      // A fragment (documented 16-bit A layout):
      //   VGPR j=0..3 <- K pairs [8*laneHi .. 8*laneHi+8)  (two aligned float4 runs)
      //   VGPR j=4..7 <- K pairs [16+8*laneHi .. 24+8*laneHi)
      const float* rowp = &ldsA[cur][(rt << 4) + laneN][0];
      float4 q0 = *(const float4*)(rowp + fo);
      float4 q1 = *(const float4*)(rowp + fo + 4);
      float4 q2 = *(const float4*)(rowp + fo + 16);
      float4 q3 = *(const float4*)(rowp + fo + 20);
      union { v2hp p[8]; v16h h; } A;
      A.p[0] = __builtin_amdgcn_cvt_pkrtz(q0.x, q0.y);
      A.p[1] = __builtin_amdgcn_cvt_pkrtz(q0.z, q0.w);
      A.p[2] = __builtin_amdgcn_cvt_pkrtz(q1.x, q1.y);
      A.p[3] = __builtin_amdgcn_cvt_pkrtz(q1.z, q1.w);
      A.p[4] = __builtin_amdgcn_cvt_pkrtz(q2.x, q2.y);
      A.p[5] = __builtin_amdgcn_cvt_pkrtz(q2.z, q2.w);
      A.p[6] = __builtin_amdgcn_cvt_pkrtz(q3.x, q3.y);
      A.p[7] = __builtin_amdgcn_cvt_pkrtz(q3.z, q3.w);
#pragma unroll
      for (int i = 0; i < 4; ++i) {
        acc[rt][i] = __builtin_amdgcn_wmma_f32_16x16x32_f16(
            false, A.h, false, bf[i], (short)0, acc[rt][i], false, false);
      }
    }

    if (kt < CDIM / 32 - 1) stage_wait();
    __syncthreads();
  }

  // epilogue: + b1 + hproj, tanh, dot with V, reduce to per-row sums
  float rsum[4][8];
#pragma unroll
  for (int rt = 0; rt < 4; ++rt)
#pragma unroll
    for (int r = 0; r < 8; ++r) rsum[rt][r] = 0.f;

#pragma unroll
  for (int i = 0; i < 4; ++i) {
    const int n  = (wave << 6) + (i << 4) + laneN;
    const float bias = b1[n] + hproj[batch * 512 + n];
    const float vn   = V[n];
#pragma unroll
    for (int rt = 0; rt < 4; ++rt)
#pragma unroll
      for (int r = 0; r < 8; ++r)
        rsum[rt][r] += HW_TANH(acc[rt][i][r] + bias) * vn;
  }

#pragma unroll
  for (int rt = 0; rt < 4; ++rt)
#pragma unroll
    for (int r = 0; r < 8; ++r) {
      float s = rsum[rt][r];
#pragma unroll
      for (int off = 1; off < 16; off <<= 1) s += __shfl_xor(s, off, 32);
      if (laneN == 0)
        atomicAdd(&ldsSum[(rt << 4) + (laneHi << 3) + r], s);
    }

  __syncthreads();
  if (tid < ROWS_BLK) score[row0 + tid] = ldsSum[tid] + bV[0];
}

// ---------------- masked softmax over 1024 positions per batch ----------------
__global__ __launch_bounds__(256) void k_softmax(const float* __restrict__ score,
                                                 const int* __restrict__ mask,
                                                 float* __restrict__ alpha) {
  __shared__ float red[256];
  const int b = blockIdx.x, t = threadIdx.x;

  float v[4];
#pragma unroll
  for (int j = 0; j < 4; ++j) {
    int idx = t + j * 256;
    float s = score[b * 1024 + idx];
    v[j] = (mask[b * 1024 + idx] > 0) ? s : NEG_INF;
  }
  float m = fmaxf(fmaxf(v[0], v[1]), fmaxf(v[2], v[3]));
  red[t] = m; __syncthreads();
  for (int off = 128; off > 0; off >>= 1) {
    if (t < off) red[t] = fmaxf(red[t], red[t + off]);
    __syncthreads();
  }
  m = red[0]; __syncthreads();

  float e[4]; float s = 0.f;
#pragma unroll
  for (int j = 0; j < 4; ++j) { e[j] = expf(v[j] - m); s += e[j]; }
  red[t] = s; __syncthreads();
  for (int off = 128; off > 0; off >>= 1) {
    if (t < off) red[t] += red[t + off];
    __syncthreads();
  }
  s = red[0];

  float inv = 1.f / s;
#pragma unroll
  for (int j = 0; j < 4; ++j) alpha[b * 1024 + t + j * 256] = e[j] * inv;
}

// ---------------- context[b,c] = sum_hw feat[b,hw,c] * alpha[b,hw] ----------------
__global__ __launch_bounds__(256) void k_context(const float* __restrict__ feat,
                                                 const float* __restrict__ alpha,
                                                 float* __restrict__ ctx) {
  const int b = blockIdx.x;
  const int c0 = threadIdx.x * 2;
  const int hw0 = blockIdx.y * 128;
  float a0 = 0.f, a1 = 0.f;
  for (int i = 0; i < 128; ++i) {
    int hw = hw0 + i;
    float al = alpha[b * 1024 + hw];
    float2 f = *(const float2*)(feat + ((size_t)(b * 1024 + hw)) * CDIM + c0);
    a0 += al * f.x;
    a1 += al * f.y;
  }
  atomicAdd(&ctx[b * CDIM + c0],     a0);
  atomicAdd(&ctx[b * CDIM + c0 + 1], a1);
}

extern "C" void kernel_launch(void* const* d_in, const int* in_sizes, int n_in,
                              void* d_out, int out_size, void* d_ws, size_t ws_size,
                              hipStream_t stream) {
  (void)in_sizes; (void)n_in; (void)out_size; (void)ws_size;

  const float* feat   = (const float*)d_in[0];   // [64,32,32,512]
  const float* hidden = (const float*)d_in[1];   // [64,512]
  const int*   mask   = (const int*)  d_in[2];   // [64,32,32]
  const float* W1     = (const float*)d_in[3];   // [512,512]
  const float* b1     = (const float*)d_in[4];   // [512]
  const float* W2     = (const float*)d_in[5];   // [512,512]
  const float* b2     = (const float*)d_in[6];   // [512]
  const float* V      = (const float*)d_in[7];   // [512,1]
  const float* bV     = (const float*)d_in[8];   // [1]

  float* out   = (float*)d_out;
  float* ctx   = out;               // [64,512]
  float* alpha = out + 64 * 512;    // [64,32,32]

  char* ws = (char*)d_ws;
  _Float16* W1p  = (_Float16*)ws;                        // 512 KB
  float*    hpr  = (float*)(ws + 524288);                // 128 KB
  float*    scr  = (float*)(ws + 524288 + 131072);       // 256 KB

  k_pack_w1<<<(CDIM * ADIM + 255) / 256, 256, 0, stream>>>(W1, W1p);
  k_hproj  <<<64, 256, 0, stream>>>(hidden, W2, b2, hpr);
  k_zero   <<<(64 * CDIM + 255) / 256, 256, 0, stream>>>(ctx, 64 * CDIM);
  k_score  <<<N_ROWS / ROWS_BLK, 256, 0, stream>>>(feat, W1p, b1, hpr, V, bV, scr);
  k_softmax<<<64, 256, 0, stream>>>(scr, mask, alpha);
  k_context<<<dim3(64, 8), 256, 0, stream>>>(feat, alpha, ctx);
}
